// ASPMSoftMasking_13700945674779
// MI455X (gfx1250) — matrix operand order repaired and checked by reading, not compiled, not run on hardware
//
#include <hip/hip_runtime.h>

typedef __attribute__((ext_vector_type(16))) __bf16 v16bf;
typedef __attribute__((ext_vector_type(8)))  __bf16 v8bf;
typedef __attribute__((ext_vector_type(8)))  float  v8f;

#define B_    16
#define T_    8192
#define D_    512
#define KMASK 5734            // int(8192 * 0.7)
#define ROWS  128             // x rows staged per workgroup (8 waves x 16)
#define PAD   8               // bf16 elements of row padding (16B -> bank skew)
#define LDW   (D_ + PAD)      // 520
#define NKT   16              // K tiles (512 / 32)
#define NNT   32              // N tiles (512 / 16)

// ---------------------------------------------------------------------------
// Kernel 1: pack W (fp32, [e][d]) into bf16 WMMA B-fragments.
// B[k][n] = W[n][k].  Fragment (nt,kt): 32 lanes x 16 bf16, lane l holds
// N = nt*16 + (l&15), K = kt*32 + 16*(l>>4) + j  (j = 0..15).
// Fragments for one nt (16 kt's) are contiguous: 16 KB per nt.
// ---------------------------------------------------------------------------
__global__ void __launch_bounds__(256) pack_w(const float* __restrict__ W,
                                              __bf16* __restrict__ Wfrag) {
  int g      = blockIdx.x * 256 + threadIdx.x;  // 0 .. 512*512-1
  int frag   = g >> 9;
  int within = g & 511;
  int l  = within >> 4;
  int j  = within & 15;
  int nt = frag >> 4;
  int kt = frag & 15;
  int e  = nt * 16 + (l & 15);
  int d  = kt * 32 + ((l >> 4) << 4) + j;
  Wfrag[g] = (__bf16)W[e * D_ + d];
}

// ---------------------------------------------------------------------------
// Kernel 2: fused GEMM + tanh + v-dot -> scores (B*T floats).
// 256 threads = 8 waves; each wave owns a 16-row M tile (A register-cached);
// B fragments staged per-nt into LDS and shared by all 8 waves.
// ---------------------------------------------------------------------------
__global__ void __launch_bounds__(256) gemm_score(const float*  __restrict__ x,
                                                  const __bf16* __restrict__ Wfrag,
                                                  const float*  __restrict__ bias,
                                                  const float*  __restrict__ vvec,
                                                  float* __restrict__ scores) {
  extern __shared__ __bf16 lds[];
  __bf16* tileA = lds;                    // ROWS * LDW bf16 = 133,120 B
  __bf16* Blds  = lds + ROWS * LDW;       // NKT * 512 bf16  =  16,384 B

  const int  tid     = threadIdx.x;
  const long rowbase = (long)blockIdx.x * ROWS;

  // ---- Stage 128x512 fp32 -> bf16 LDS (padded rows) ----
  const float4* x4 = (const float4*)(x + rowbase * D_);
  for (int s = tid; s < ROWS * (D_ / 4); s += 256) {
    int r  = s >> 7;          // / 128
    int c4 = s & 127;
    float4 v = x4[r * (D_ / 4) + c4];
    int o = r * LDW + c4 * 4;
    tileA[o + 0] = (__bf16)v.x; tileA[o + 1] = (__bf16)v.y;
    tileA[o + 2] = (__bf16)v.z; tileA[o + 3] = (__bf16)v.w;
  }
  __syncthreads();

  const int wv   = tid >> 5;
  const int lane = tid & 31;
  const int mloc = lane & 15;   // N column inside C tile / M row inside A tile
  const int hh   = lane >> 4;   // half select
  const int rowA = wv * 16 + mloc;

  // ---- Register-cache this wave's 16 A-fragments (256 bf16 per lane) ----
  v16bf afrag[NKT];
#pragma unroll
  for (int kt = 0; kt < NKT; ++kt) {
    int k0 = kt * 32 + hh * 16;
    v8bf lo = *(const v8bf*)&tileA[rowA * LDW + k0];        // K = k0..k0+7
    v8bf hi = *(const v8bf*)&tileA[rowA * LDW + k0 + 8];    // K = k0+8..k0+15
    afrag[kt] = __builtin_shufflevector(lo, hi, 0, 1, 2, 3, 4, 5, 6, 7,
                                                8, 9, 10, 11, 12, 13, 14, 15);
  }

  float partial[8] = {0.f, 0.f, 0.f, 0.f, 0.f, 0.f, 0.f, 0.f};

  for (int nt = 0; nt < NNT; ++nt) {
    // ---- Cooperative stage of this nt's 16 B-fragments (16 KB) to LDS ----
    __syncthreads();                       // previous nt's Blds reads complete
    {
      const float4* src = (const float4*)(Wfrag + (size_t)nt * NKT * 512);
      float4*       dst = (float4*)Blds;   // 1024 float4 total
      dst[tid]       = src[tid];
      dst[tid + 256] = src[tid + 256];
      dst[tid + 512] = src[tid + 512];
      dst[tid + 768] = src[tid + 768];
    }
    __syncthreads();

    v8f acc = {};
#pragma unroll
    for (int kt = 0; kt < NKT; ++kt) {
      v16bf bm = *(const v16bf*)&Blds[kt * 512 + lane * 16];
      acc = __builtin_amdgcn_wmma_f32_16x16x32_bf16(
          /*neg_a=*/false, afrag[kt], /*neg_b=*/false, bm,
          /*c_mod=*/(short)0, acc, /*reuse_a=*/false, /*reuse_b=*/false);
    }

    // Epilogue: this lane holds h[M = r + 8*hh][N = mloc] in acc[r].
    int   e  = nt * 16 + mloc;
    float be = bias[e];
    float ve = vvec[e];
#pragma unroll
    for (int r = 0; r < 8; ++r)
      partial[r] += tanhf(acc[r] + be) * ve;
  }

  // Reduce across the 16 lanes of each half (N dimension).
#pragma unroll
  for (int r = 0; r < 8; ++r) {
    float p = partial[r];
    p += __shfl_xor(p, 1, 32);
    p += __shfl_xor(p, 2, 32);
    p += __shfl_xor(p, 4, 32);
    p += __shfl_xor(p, 8, 32);
    partial[r] = p;
  }
  if (mloc == 0) {
    long base = rowbase + wv * 16 + hh * 8;
#pragma unroll
    for (int r = 0; r < 8; ++r) scores[base + r] = partial[r];
  }
}

// ---------------------------------------------------------------------------
// Kernel 3: per-batch softmax + radix-select threshold + stable-tie mask.
// One 256-thread block per batch row. Writes maw (B*T floats).
// ---------------------------------------------------------------------------
__device__ __forceinline__ unsigned keyOf(float f) {
  unsigned u = __float_as_uint(f);
  return (u & 0x80000000u) ? ~u : (u ^ 0x80000000u);  // order-preserving
}

__global__ void __launch_bounds__(256) select_softmax(const float* __restrict__ scores,
                                                      float* __restrict__ maw) {
  __shared__ float    sc[T_];          // 32 KB
  __shared__ unsigned hist[256];
  __shared__ float    red[8];
  __shared__ unsigned ured[8];
  __shared__ float    s_max, s_den;
  __shared__ unsigned s_prefix, s_remaining, s_below;

  const int b    = blockIdx.x;
  const int tid  = threadIdx.x;
  const int lane = tid & 31;
  const int wv   = tid >> 5;
  const float* srow = scores + (long)b * T_;

  // ---- block max ----
  float lmax = -3.4e38f;
  for (int t = tid; t < T_; t += 256) { float s = srow[t]; sc[t] = s; lmax = fmaxf(lmax, s); }
  for (int m = 16; m; m >>= 1) lmax = fmaxf(lmax, __shfl_xor(lmax, m, 32));
  if (lane == 0) red[wv] = lmax;
  __syncthreads();
  if (tid == 0) { float m = red[0]; for (int i = 1; i < 8; ++i) m = fmaxf(m, red[i]); s_max = m; }
  __syncthreads();
  const float mx = s_max;

  // ---- block sum of exp ----
  float lsum = 0.f;
  for (int t = tid; t < T_; t += 256) lsum += expf(sc[t] - mx);
  for (int m = 16; m; m >>= 1) lsum += __shfl_xor(lsum, m, 32);
  if (lane == 0) red[wv] = lsum;
  __syncthreads();
  if (tid == 0) { float s = 0.f; for (int i = 0; i < 8; ++i) s += red[i]; s_den = s; }
  __syncthreads();
  const float inv_den = 1.0f / s_den;

  // ---- radix select: key at 0-based ascending rank (KMASK-1) ----
  if (tid == 0) { s_prefix = 0u; s_remaining = KMASK - 1; }
  __syncthreads();
  for (int pass = 0; pass < 4; ++pass) {
    const int shift = 24 - 8 * pass;
    const unsigned pmask = (pass == 0) ? 0u : (0xFFFFFFFFu << (shift + 8));
    hist[tid] = 0;
    __syncthreads();
    const unsigned pref = s_prefix;
    for (int t = tid; t < T_; t += 256) {
      unsigned k = keyOf(sc[t]);
      if ((k & pmask) == (pref & pmask)) atomicAdd(&hist[(k >> shift) & 255], 1u);
    }
    __syncthreads();
    if (tid == 0) {
      unsigned rem = s_remaining, cum = 0; int bidx = 255;
      for (int i = 0; i < 256; ++i) {
        unsigned c = hist[i];
        if (cum + c > rem) { bidx = i; break; }
        cum += c;
      }
      s_prefix    = pref | ((unsigned)bidx << shift);
      s_remaining = rem - cum;
    }
    __syncthreads();
  }
  const unsigned thr = s_prefix;

  // ---- count strictly below threshold ----
  unsigned lb = 0;
  for (int t = tid; t < T_; t += 256) if (keyOf(sc[t]) < thr) lb++;
  for (int m = 16; m; m >>= 1) lb += __shfl_xor(lb, m, 32);
  if (lane == 0) ured[wv] = lb;
  __syncthreads();
  if (tid == 0) { unsigned s = 0; for (int i = 0; i < 8; ++i) s += ured[i]; s_below = s; }
  __syncthreads();
  const unsigned need = (unsigned)KMASK - s_below;  // ties to mask, index order

  float* mrow = maw + (long)b * T_;
  // Non-tie elements: fully parallel.
  for (int t = tid; t < T_; t += 256) {
    unsigned k = keyOf(sc[t]);
    if (k != thr) mrow[t] = (k < thr) ? 0.0f : expf(sc[t] - mx) * inv_den;
  }
  // Tie elements: wave 0 walks T in index order (stable argsort tie-break).
  if (wv == 0) {
    unsigned running = 0;
    for (int t0 = 0; t0 < T_; t0 += 32) {
      int  t  = t0 + lane;
      bool eq = (keyOf(sc[t]) == thr);
      unsigned long long bal = __ballot(eq);
      unsigned before = (unsigned)__popcll(bal & ((1ull << lane) - 1ull)) + running;
      if (eq) mrow[t] = (before < need) ? 0.0f : expf(sc[t] - mx) * inv_den;
      running += (unsigned)__popcll(bal);
    }
  }
}

// ---------------------------------------------------------------------------
// Kernel 4: out = x*w + (1-w)*H0, float4 streaming.
// ---------------------------------------------------------------------------
__global__ void __launch_bounds__(256) apply_mask(const float* __restrict__ x,
                                                  const float* __restrict__ H0,
                                                  const float* __restrict__ maw,
                                                  float* __restrict__ out) {
  long i = (long)blockIdx.x * 256 + threadIdx.x;  // float4 index
  const float4* x4 = (const float4*)x;
  const float4* h4 = (const float4*)H0;
  float4*       o4 = (float4*)out;
  long row = i >> 7;            // / (D_/4)
  int  c   = (int)(i & 127);
  float w  = maw[row];
  float iw = 1.0f - w;
  float4 xv = x4[i];
  float4 hv = h4[c];
  float4 r;
  r.x = xv.x * w + iw * hv.x;
  r.y = xv.y * w + iw * hv.y;
  r.z = xv.z * w + iw * hv.z;
  r.w = xv.w * w + iw * hv.w;
  o4[i] = r;
}

// ---------------------------------------------------------------------------
extern "C" void kernel_launch(void* const* d_in, const int* in_sizes, int n_in,
                              void* d_out, int out_size, void* d_ws, size_t ws_size,
                              hipStream_t stream) {
  const float* x  = (const float*)d_in[0];
  const float* W  = (const float*)d_in[1];
  const float* b  = (const float*)d_in[2];
  const float* v  = (const float*)d_in[3];
  // d_in[4] = vb: a per-batch constant shift -> cancels in softmax and argsort.
  const float* H0 = (const float*)d_in[5];

  float* out = (float*)d_out;
  float* maw = out + (long)B_ * T_ * D_;          // second output, tail of d_out

  __bf16* Wfrag  = (__bf16*)d_ws;                                  // 512 KB
  float*  scores = (float*)((char*)d_ws + (size_t)D_ * D_ * sizeof(__bf16));

  pack_w<<<(D_ * D_) / 256, 256, 0, stream>>>(W, Wfrag);

  size_t lds_bytes = (size_t)(ROWS * LDW + NKT * 512) * sizeof(__bf16);  // 149,504
  gemm_score<<<(B_ * T_) / ROWS, 256, lds_bytes, stream>>>(x, Wfrag, b, v, scores);

  select_softmax<<<B_, 256, 0, stream>>>(scores, maw);

  apply_mask<<<(long)(B_ * T_) * (D_ / 4) / 256, 256, 0, stream>>>(x, H0, maw, out);
}
